// Matching_34402688041067
// MI455X (gfx1250) — compile-verified, matrix operand order checked by reading, not run on hardware
//
#include <hip/hip_runtime.h>
#include <stdint.h>

typedef __attribute__((ext_vector_type(16))) __bf16 bf16x16;
typedef __attribute__((ext_vector_type(8)))  float  f32x8;

union FragB { uint4 q[2]; bf16x16 v; };

#define NBATCH 4
#define LDIM 4096
#define SDIM 4096
#define CDIM 256
#define NL (NBATCH*LDIM)
#define NS (NBATCH*SDIM)
static const long long NLS = (long long)NBATCH * LDIM * SDIM;

__device__ __forceinline__ unsigned short f2bf(float f) {
  unsigned u = __float_as_uint(f);
  u += 0x7FFFu + ((u >> 16) & 1u);         // round-to-nearest-even
  return (unsigned short)(u >> 16);
}
__device__ __forceinline__ float bf2f(unsigned short h) {
  return __uint_as_float(((unsigned)h) << 16);
}

// ---------------------------------------------------------------------------
// K1: scale by 1/sqrt(C)=1/16, split f32 -> bf16 hi/lo, row squared norms.
// One wave per feature row (256 f32). grid = 2*NL/8 blocks of 256.
// ---------------------------------------------------------------------------
__global__ __launch_bounds__(256) void k_convert(
    const float* __restrict__ f0, const float* __restrict__ f1,
    unsigned short* __restrict__ f0h, unsigned short* __restrict__ f0l,
    unsigned short* __restrict__ f1h, unsigned short* __restrict__ f1l,
    float* __restrict__ sq0, float* __restrict__ sq1) {
  const int wv = threadIdx.x >> 5, lane = threadIdx.x & 31;
  const int row = blockIdx.x * 8 + wv;               // 0 .. 2*NL-1
  const float* src; unsigned short *dh, *dl; float* sq; int rr;
  if (row < NL) { src = f0; dh = f0h; dl = f0l; sq = sq0; rr = row; }
  else          { src = f1; dh = f1h; dl = f1l; sq = sq1; rr = row - NL; }

  const float4* p = (const float4*)(src + (size_t)rr * CDIM + lane * 8);
  float4 a = p[0], b = p[1];
  float x[8] = {a.x, a.y, a.z, a.w, b.x, b.y, b.z, b.w};
  unsigned hu[4], lu[4];
  float ss = 0.f;
#pragma unroll
  for (int i = 0; i < 4; ++i) {
    float x0 = x[2*i] * 0.0625f, x1 = x[2*i+1] * 0.0625f;
    ss += x0 * x0 + x1 * x1;
    unsigned short h0 = f2bf(x0), h1 = f2bf(x1);
    unsigned short l0 = f2bf(x0 - bf2f(h0)), l1 = f2bf(x1 - bf2f(h1));
    hu[i] = (unsigned)h0 | ((unsigned)h1 << 16);
    lu[i] = (unsigned)l0 | ((unsigned)l1 << 16);
  }
  *(uint4*)(dh + (size_t)rr * CDIM + lane * 8) = make_uint4(hu[0], hu[1], hu[2], hu[3]);
  *(uint4*)(dl + (size_t)rr * CDIM + lane * 8) = make_uint4(lu[0], lu[1], lu[2], lu[3]);
#pragma unroll
  for (int o = 16; o > 0; o >>= 1) ss += __shfl_xor(ss, o, 32);
  if (lane == 0) sq[rr] = ss;
}

// ---------------------------------------------------------------------------
// K2: WMMA GEMM (bf16 hi/lo split) + sim epilogue.
// Block 256 (8 waves) computes a 128x128 tile. LDS stages A/B hi/lo per
// 32-wide K chunk. sim = -sqrt(max(sq0+sq1-2*inner,0)) / (0.1*256).
// ---------------------------------------------------------------------------
__global__ __launch_bounds__(256) void k_gemm_sim(
    const unsigned short* __restrict__ f0h, const unsigned short* __restrict__ f0l,
    const unsigned short* __restrict__ f1h, const unsigned short* __restrict__ f1l,
    const float* __restrict__ sq0, const float* __restrict__ sq1,
    float* __restrict__ simOut) {
  __shared__ unsigned short Ah[128][48], Al[128][48], Bh[128][48], Bl[128][48];
  const int tn = blockIdx.x, tm = blockIdx.y, n = blockIdx.z;
  const int t = threadIdx.x;
  const int lane = t & 31, wv = t >> 5;
  const int half = lane >> 4, nn = lane & 15;

  const size_t arow = (size_t)(n * LDIM + tm * 128) * CDIM;
  const size_t brow = (size_t)(n * SDIM + tn * 128) * CDIM;

  f32x8 acc[8];
#pragma unroll
  for (int j = 0; j < 8; ++j)
#pragma unroll
    for (int e = 0; e < 8; ++e) acc[j][e] = 0.f;

  for (int kc = 0; kc < CDIM / 32; ++kc) {
    __syncthreads();
#pragma unroll
    for (int p = 0; p < 2; ++p) {
      int idx = t + p * 256;                 // 512 uint4 per array
      int r = idx >> 2, sg = (idx & 3) * 8;  // row, k-offset (in ushorts)
      size_t go = (size_t)r * CDIM + kc * 32 + sg;
      *(uint4*)&Ah[r][sg] = *(const uint4*)(f0h + arow + go);
      *(uint4*)&Al[r][sg] = *(const uint4*)(f0l + arow + go);
      *(uint4*)&Bh[r][sg] = *(const uint4*)(f1h + brow + go);
      *(uint4*)&Bl[r][sg] = *(const uint4*)(f1l + brow + go);
    }
    __syncthreads();
    // A fragment (16x32 bf16, ISA 7.12.2 layout)
    const int mrow = wv * 16 + nn;
    const int k0 = half * 8, k1 = 16 + half * 8;
    FragB ah, al;
    ah.q[0] = *(const uint4*)&Ah[mrow][k0];
    ah.q[1] = *(const uint4*)&Ah[mrow][k1];
    al.q[0] = *(const uint4*)&Al[mrow][k0];
    al.q[1] = *(const uint4*)&Al[mrow][k1];
#pragma unroll
    for (int j = 0; j < 8; ++j) {
      const int br = j * 16 + nn;            // B column = f1 row
      FragB bh, bl;
      bh.q[0] = *(const uint4*)&Bh[br][k0];
      bh.q[1] = *(const uint4*)&Bh[br][k1];
      bl.q[0] = *(const uint4*)&Bl[br][k0];
      bl.q[1] = *(const uint4*)&Bl[br][k1];
      acc[j] = __builtin_amdgcn_wmma_f32_16x16x32_bf16(false, ah.v, false, bh.v,
                                                       (short)0, acc[j], false, false);
      acc[j] = __builtin_amdgcn_wmma_f32_16x16x32_bf16(false, ah.v, false, bl.v,
                                                       (short)0, acc[j], false, false);
      acc[j] = __builtin_amdgcn_wmma_f32_16x16x32_bf16(false, al.v, false, bh.v,
                                                       (short)0, acc[j], false, false);
    }
  }

  float sq0r[8];
#pragma unroll
  for (int e = 0; e < 8; ++e)
    sq0r[e] = sq0[n * LDIM + tm * 128 + wv * 16 + e + 8 * half];
#pragma unroll
  for (int j = 0; j < 8; ++j) {
    const int gn = tn * 128 + j * 16 + nn;
    const float sq1v = sq1[n * SDIM + gn];
#pragma unroll
    for (int e = 0; e < 8; ++e) {
      const int gm = tm * 128 + wv * 16 + e + 8 * half;   // C/D: M = e + 8*half
      float d2 = fmaxf(sq0r[e] + sq1v - 2.f * acc[j][e], 0.f);
      simOut[(size_t)(n * LDIM + gm) * SDIM + gn] = -__fsqrt_rn(d2) * 0.0390625f;
    }
  }
}

// ---------------------------------------------------------------------------
// K3: per 16-row band: full row exp-sums (owned, tree-reduced, deterministic)
// and per-band column partial sums.  grid = NBATCH*LDIM/16.
// ---------------------------------------------------------------------------
__global__ __launch_bounds__(256) void k_rowsum_colpart(
    const float* __restrict__ simIn, float* __restrict__ rowsum,
    float* __restrict__ colpart) {
  __shared__ float red[256];
  const int band = blockIdx.x, n = band >> 8, lb = band & 255;
  const int t = threadIdx.x;
  float4 cp[4];
#pragma unroll
  for (int i = 0; i < 4; ++i) cp[i] = make_float4(0.f, 0.f, 0.f, 0.f);
  for (int r = 0; r < 16; ++r) {
    const int l = lb * 16 + r;
    const float4* srow = (const float4*)(simIn + (size_t)(n * LDIM + l) * SDIM);
    float rs = 0.f;
#pragma unroll
    for (int i = 0; i < 4; ++i) {
      float4 v = srow[i * 256 + t];
      float4 e = make_float4(expf(v.x), expf(v.y), expf(v.z), expf(v.w));
      cp[i].x += e.x; cp[i].y += e.y; cp[i].z += e.z; cp[i].w += e.w;
      rs += e.x + e.y + e.z + e.w;
    }
    red[t] = rs; __syncthreads();
    for (int o = 128; o > 0; o >>= 1) { if (t < o) red[t] += red[t + o]; __syncthreads(); }
    if (t == 0) rowsum[n * LDIM + l] = red[0];
    __syncthreads();
  }
  float4* cpo = (float4*)(colpart + (size_t)(n * 256 + lb) * SDIM);
#pragma unroll
  for (int i = 0; i < 4; ++i) cpo[i * 256 + t] = cp[i];
}

// K3c: reduce the 256 band-partials per column, fixed order (deterministic).
__global__ __launch_bounds__(256) void k_colsum(
    const float* __restrict__ colpart, float* __restrict__ colsum) {
  const int g = blockIdx.x * 256 + threadIdx.x;    // 0..NS-1
  const int n = g >> 12, s = g & (SDIM - 1);
  float acc = 0.f;
  for (int lb = 0; lb < 256; ++lb)
    acc += colpart[(size_t)(n * 256 + lb) * SDIM + s];
  colsum[g] = acc;
}

// ---------------------------------------------------------------------------
// K4: conf0 = e/colsum, conf1 = e/rowsum, conf = conf0*conf1.
// Row conf-max stored directly (band owns rows); column conf-max via uint
// atomicMax on the bit pattern (conf >= 0 -> IEEE order == integer order).
// ---------------------------------------------------------------------------
__global__ __launch_bounds__(256) void k_conf(
    const float* __restrict__ simIn, const float* __restrict__ rowsum,
    const float* __restrict__ colsum,
    float* __restrict__ conf, float* __restrict__ conf0, float* __restrict__ conf1,
    float* __restrict__ rowmaxc, unsigned* __restrict__ colmaxc) {
  __shared__ float red[256];
  const int band = blockIdx.x, n = band >> 8, lb = band & 255;
  const int t = threadIdx.x;
  float4 icv[4], cm[4];
#pragma unroll
  for (int i = 0; i < 4; ++i) {
    float4 cs = ((const float4*)(colsum + (size_t)n * SDIM))[i * 256 + t];
    icv[i] = make_float4(1.f / cs.x, 1.f / cs.y, 1.f / cs.z, 1.f / cs.w);
    cm[i] = make_float4(0.f, 0.f, 0.f, 0.f);
  }
  for (int r = 0; r < 16; ++r) {
    const int l = lb * 16 + r;
    const size_t rowoff = (size_t)(n * LDIM + l) * SDIM;
    const float4* srow = (const float4*)(simIn + rowoff);
    float4* cO  = (float4*)(conf  + rowoff);
    float4* c0O = (float4*)(conf0 + rowoff);
    float4* c1O = (float4*)(conf1 + rowoff);
    const float ir = 1.f / rowsum[n * LDIM + l];
    float rm = 0.f;
#pragma unroll
    for (int i = 0; i < 4; ++i) {
      float4 v = srow[i * 256 + t];
      float4 e = make_float4(expf(v.x), expf(v.y), expf(v.z), expf(v.w));
      float4 a = make_float4(e.x * icv[i].x, e.y * icv[i].y, e.z * icv[i].z, e.w * icv[i].w);
      float4 b = make_float4(e.x * ir, e.y * ir, e.z * ir, e.w * ir);
      float4 cf = make_float4(a.x * b.x, a.y * b.y, a.z * b.z, a.w * b.w);
      c0O[i * 256 + t] = a; c1O[i * 256 + t] = b; cO[i * 256 + t] = cf;
      cm[i].x = fmaxf(cm[i].x, cf.x); cm[i].y = fmaxf(cm[i].y, cf.y);
      cm[i].z = fmaxf(cm[i].z, cf.z); cm[i].w = fmaxf(cm[i].w, cf.w);
      rm = fmaxf(rm, fmaxf(fmaxf(cf.x, cf.y), fmaxf(cf.z, cf.w)));
    }
    red[t] = rm; __syncthreads();
    for (int o = 128; o > 0; o >>= 1) { if (t < o) red[t] = fmaxf(red[t], red[t + o]); __syncthreads(); }
    if (t == 0) rowmaxc[n * LDIM + l] = red[0];
    __syncthreads();
  }
#pragma unroll
  for (int i = 0; i < 4; ++i) {
    const int c = n * SDIM + i * 1024 + t * 4;
    atomicMax(&colmaxc[c + 0], __float_as_uint(cm[i].x));
    atomicMax(&colmaxc[c + 1], __float_as_uint(cm[i].y));
    atomicMax(&colmaxc[c + 2], __float_as_uint(cm[i].z));
    atomicMax(&colmaxc[c + 3], __float_as_uint(cm[i].w));
  }
}

// ---------------------------------------------------------------------------
// K5: mutual-NN + threshold, per row. argmax(mask) = first matching index.
// ---------------------------------------------------------------------------
__global__ __launch_bounds__(256) void k_match(
    const float* __restrict__ conf, const float* __restrict__ rowmaxc,
    const float* __restrict__ colmaxcF,
    float* __restrict__ maskv, float* __restrict__ jids, float* __restrict__ mconf) {
  __shared__ int red[256];
  const int row = blockIdx.x;                      // 0..NL-1
  const int n = row >> 12;
  const int t = threadIdx.x;
  const float rm = rowmaxc[row];
  const float4* crow = (const float4*)(conf + (size_t)row * SDIM);
  const float4* cmax = (const float4*)(colmaxcF + (size_t)n * SDIM);
  int best = 0x7FFFFFFF;
#pragma unroll
  for (int i = 0; i < 4; ++i) {
    float4 v = crow[i * 256 + t];
    float4 m = cmax[i * 256 + t];
    const int c = i * 1024 + t * 4;
    if (v.x > 0.2f && v.x == rm && v.x == m.x) best = min(best, c + 0);
    if (v.y > 0.2f && v.y == rm && v.y == m.y) best = min(best, c + 1);
    if (v.z > 0.2f && v.z == rm && v.z == m.z) best = min(best, c + 2);
    if (v.w > 0.2f && v.w == rm && v.w == m.w) best = min(best, c + 3);
  }
  red[t] = best; __syncthreads();
  for (int o = 128; o > 0; o >>= 1) { if (t < o) red[t] = min(red[t], red[t + o]); __syncthreads(); }
  if (t == 0) {
    const int j = red[0];
    const bool found = (j != 0x7FFFFFFF);
    maskv[row] = found ? 1.f : 0.f;
    jids[row]  = found ? (float)j : 0.f;
    mconf[row] = found ? rm : 0.f;    // conf at argmax == rowmax by construction
  }
}

// ---------------------------------------------------------------------------
extern "C" void kernel_launch(void* const* d_in, const int* in_sizes, int n_in,
                              void* d_out, int out_size, void* d_ws, size_t ws_size,
                              hipStream_t stream) {
  const float* f0 = (const float*)d_in[0];
  const float* f1 = (const float*)d_in[1];
  float* out = (float*)d_out;

  char* ws = (char*)d_ws;
  const size_t SZ_BF = (size_t)NL * CDIM * 2;          // 8 MiB per bf16 array
  unsigned short* f0h = (unsigned short*)(ws);
  unsigned short* f0l = (unsigned short*)(ws + 1 * SZ_BF);
  unsigned short* f1h = (unsigned short*)(ws + 2 * SZ_BF);
  unsigned short* f1l = (unsigned short*)(ws + 3 * SZ_BF);
  char* p = ws + 4 * SZ_BF;
  float* sq0     = (float*)p; p += (size_t)NL * 4;
  float* sq1     = (float*)p; p += (size_t)NS * 4;
  float* rowsum  = (float*)p; p += (size_t)NL * 4;
  float* colsum  = (float*)p; p += (size_t)NS * 4;
  float* rowmaxc = (float*)p; p += (size_t)NL * 4;
  float* colmaxc = (float*)p; p += (size_t)NS * 4;
  float* colpart = (float*)p;                          // NBATCH*256*SDIM*4 = 16 MiB

  float* conf  = out;
  float* conf0 = out + NLS;
  float* conf1 = out + 2 * NLS;
  float* simO  = out + 3 * NLS;
  float* maskv = out + 4 * NLS;
  float* jids  = maskv + NL;
  float* mconf = jids + NL;

  hipMemsetAsync(colmaxc, 0, (size_t)NS * 4, stream);  // atomic-max init

  k_convert<<<2 * NL / 8, 256, 0, stream>>>(f0, f1, f0h, f0l, f1h, f1l, sq0, sq1);
  k_gemm_sim<<<dim3(SDIM / 128, LDIM / 128, NBATCH), 256, 0, stream>>>(
      f0h, f0l, f1h, f1l, sq0, sq1, simO);
  k_rowsum_colpart<<<NBATCH * LDIM / 16, 256, 0, stream>>>(simO, rowsum, colpart);
  k_colsum<<<NS / 256, 256, 0, stream>>>(colpart, colsum);
  k_conf<<<NBATCH * LDIM / 16, 256, 0, stream>>>(simO, rowsum, colsum,
                                                 conf, conf0, conf1,
                                                 rowmaxc, (unsigned*)colmaxc);
  k_match<<<NL, 256, 0, stream>>>(conf, rowmaxc, colmaxc, maskv, jids, mconf);
}
